// Attention_8667244003617
// MI455X (gfx1250) — compile-verified, matrix operand order checked by reading, not run on hardware
//
#include <hip/hip_runtime.h>
#include <hip/hip_bf16.h>

typedef __attribute__((ext_vector_type(16))) _Float16 v16h;
typedef __attribute__((ext_vector_type(8)))  _Float16 v8h;
typedef __attribute__((ext_vector_type(8)))  float    v8f;

#define S_LEN   4096
#define D_IN    512
#define NH      8
#define DHEAD   64
#define D_OUT   512
#define LOG2E   1.4426950408889634f

// Load a 16-wide-half slice forming one lane's share of a 16x32 f16 WMMA tile.
// p must already include the lane's (lane>>4)*8 k-offset; hi half is at +16.
__device__ __forceinline__ v16h load_tile16(const _Float16* p) {
    v8h lo = *(const v8h*)p;
    v8h hi = *(const v8h*)(p + 16);
    v16h r;
#pragma unroll
    for (int i = 0; i < 8; ++i) { r[i] = lo[i]; r[i + 8] = hi[i]; }
    return r;
}

__device__ __forceinline__ v16h load_tile16_lds(const _Float16* p) {
    v8h lo = *(const v8h*)p;
    v8h hi = *(const v8h*)(p + 16);
    v16h r;
#pragma unroll
    for (int i = 0; i < 8; ++i) { r[i] = lo[i]; r[i + 8] = hi[i]; }
    return r;
}

// ---------------- conversion kernels ----------------

__global__ void cvt_x_kernel(const float* __restrict__ x, _Float16* __restrict__ xh, int n) {
    int i = blockIdx.x * blockDim.x + threadIdx.x;
    if (i < n) xh[i] = (_Float16)x[i];
}

// W: [H][512][64] row-major  ->  Wt: [H][64][512]  (Wt[h][n][k] = W[h][k][n] * scale)
__global__ void cvt_w_kernel(const float* __restrict__ W, _Float16* __restrict__ Wt, float scale) {
    int i = blockIdx.x * blockDim.x + threadIdx.x;   // over 8*512*64
    int h = i >> 15;
    int rem = i & 32767;
    int k = rem >> 6;
    int n = rem & 63;
    Wt[(((size_t)h * 64) + n) * 512 + k] = (_Float16)(W[i] * scale);
}

// Wo: [H*64][512] row-major -> Wot: [512][H*64]   (Wot[n][kk] = Wo[kk][n])
__global__ void cvt_wo_kernel(const float* __restrict__ Wo, _Float16* __restrict__ Wot) {
    int i = blockIdx.x * blockDim.x + threadIdx.x;   // over 512*512
    int kk = i >> 9;
    int n = i & 511;
    Wot[(size_t)n * 512 + kk] = (_Float16)Wo[i];
}

// ---------------- QKV projection ----------------
// grid (64, 8), block 128.  wave -> one 16-row s-tile of one head.
// Qh, Kh: [H][S][64]  ;  Vt: [H][64][S]
__global__ __launch_bounds__(128) void qkv_proj_kernel(
    const _Float16* __restrict__ xh,
    const _Float16* __restrict__ Wqt, const _Float16* __restrict__ Wkt,
    const _Float16* __restrict__ Wvt,
    _Float16* __restrict__ Qh, _Float16* __restrict__ Kh, _Float16* __restrict__ Vt) {

    const int lane = threadIdx.x & 31;
    const int wave = threadIdx.x >> 5;
    const int stile = blockIdx.x * 4 + wave;   // 0..255
    const int h = blockIdx.y;                  // 0..7
    const int s0 = stile * 16;
    const int ln = lane & 15;
    const int lh8 = (lane >> 4) * 8;

    v8f accQ[4] = {}, accK[4] = {}, accV[4] = {};

    const _Float16* xrow = xh + (size_t)(s0 + ln) * D_IN;
    const _Float16* wqb = Wqt + (size_t)h * DHEAD * D_IN;
    const _Float16* wkb = Wkt + (size_t)h * DHEAD * D_IN;
    const _Float16* wvb = Wvt + (size_t)h * DHEAD * D_IN;

    for (int kb = 0; kb < D_IN; kb += 32) {
        v16h a = load_tile16(xrow + kb + lh8);
#pragma unroll
        for (int t = 0; t < 4; ++t) {
            const size_t boff = (size_t)(t * 16 + ln) * D_IN + kb + lh8;
            v16h bq = load_tile16(wqb + boff);
            v16h bk = load_tile16(wkb + boff);
            v16h bv = load_tile16(wvb + boff);
            accQ[t] = __builtin_amdgcn_wmma_f32_16x16x32_f16(false, a, false, bq, (short)0, accQ[t], false, false);
            accK[t] = __builtin_amdgcn_wmma_f32_16x16x32_f16(false, a, false, bk, (short)0, accK[t], false, false);
            accV[t] = __builtin_amdgcn_wmma_f32_16x16x32_f16(false, a, false, bv, (short)0, accV[t], false, false);
        }
    }

#pragma unroll
    for (int t = 0; t < 4; ++t) {
        const int n = t * 16 + ln;
#pragma unroll
        for (int r = 0; r < 8; ++r) {
            const int m = r + lh8;             // C/D layout: m = vgpr + 8*(lane>=16)
            Qh[((size_t)h * S_LEN + s0 + m) * DHEAD + n] = (_Float16)accQ[t][r];
            Kh[((size_t)h * S_LEN + s0 + m) * DHEAD + n] = (_Float16)accK[t][r];
        }
        v8h vv;
#pragma unroll
        for (int r = 0; r < 8; ++r) vv[r] = (_Float16)accV[t][r];
        *(v8h*)(Vt + ((size_t)h * DHEAD + n) * S_LEN + s0 + lh8) = vv;
    }
}

// ---------------- flash attention ----------------
// grid (64, 8), block 128.  wave -> one 16-query tile of one head.
// Oh: [S][H*64] f16
__global__ __launch_bounds__(128) void attn_kernel(
    const _Float16* __restrict__ Qh, const _Float16* __restrict__ Kh,
    const _Float16* __restrict__ Vt, _Float16* __restrict__ Oh) {

    __shared__ _Float16 Pbuf[4][16 * 32];

    const int lane = threadIdx.x & 31;
    const int wave = threadIdx.x >> 5;
    const int stile = blockIdx.x * 4 + wave;
    const int h = blockIdx.y;
    const int s0 = stile * 16;
    const int ln = lane & 15;
    const int lh8 = (lane >> 4) * 8;

    // Q A-tiles (16 x 64 = two 16x32 tiles), kept in registers
    v16h qa[2];
    const _Float16* qrow = Qh + ((size_t)h * S_LEN + s0 + ln) * DHEAD;
#pragma unroll
    for (int kk = 0; kk < 2; ++kk) qa[kk] = load_tile16(qrow + kk * 32 + lh8);

    float mrun[8], lrun[8];
    v8f accO[4] = {};
#pragma unroll
    for (int r = 0; r < 8; ++r) { mrun[r] = -1e30f; lrun[r] = 0.0f; }

    _Float16* pb = Pbuf[wave];
    const int jend = s0 + 16;

    for (int j0 = 0; j0 < jend; j0 += 32) {
        // ---- scores: Q (16x64) x K^T (64x32) -> 2 C/D tiles ----
        v8f sc[2] = {};
#pragma unroll
        for (int nt = 0; nt < 2; ++nt) {
            const _Float16* krow = Kh + ((size_t)h * S_LEN + j0 + nt * 16 + ln) * DHEAD;
#pragma unroll
            for (int kk = 0; kk < 2; ++kk) {
                v16h kb = load_tile16(krow + kk * 32 + lh8);
                sc[nt] = __builtin_amdgcn_wmma_f32_16x16x32_f16(false, qa[kk], false, kb, (short)0, sc[nt], false, false);
            }
        }

        // ---- causal mask + online softmax (rows live across 16-lane halves) ----
        float rmax[8];
#pragma unroll
        for (int r = 0; r < 8; ++r) {
            const int qg = s0 + r + lh8;
#pragma unroll
            for (int nt = 0; nt < 2; ++nt) {
                const int key = j0 + nt * 16 + ln;
                float v = sc[nt][r];
                if (key > qg) v = -1e30f;
                sc[nt][r] = v;
            }
            rmax[r] = fmaxf(sc[0][r], sc[1][r]);
        }
#pragma unroll
        for (int off = 1; off < 16; off <<= 1)
#pragma unroll
            for (int r = 0; r < 8; ++r)
                rmax[r] = fmaxf(rmax[r], __shfl_xor(rmax[r], off, 32));

        float alpha[8], pf[2][8], psum[8];
#pragma unroll
        for (int r = 0; r < 8; ++r) {
            const float nm = fmaxf(mrun[r], rmax[r]);
            alpha[r] = exp2f((mrun[r] - nm) * LOG2E);
            mrun[r] = nm;
        }
#pragma unroll
        for (int r = 0; r < 8; ++r) {
            pf[0][r] = exp2f((sc[0][r] - mrun[r]) * LOG2E);
            pf[1][r] = exp2f((sc[1][r] - mrun[r]) * LOG2E);
            psum[r] = pf[0][r] + pf[1][r];
        }
#pragma unroll
        for (int off = 1; off < 16; off <<= 1)
#pragma unroll
            for (int r = 0; r < 8; ++r)
                psum[r] += __shfl_xor(psum[r], off, 32);
#pragma unroll
        for (int r = 0; r < 8; ++r) lrun[r] = lrun[r] * alpha[r] + psum[r];
#pragma unroll
        for (int t = 0; t < 4; ++t)
#pragma unroll
            for (int r = 0; r < 8; ++r) accO[t][r] *= alpha[r];

        // ---- P: C/D layout -> LDS -> A layout ----
#pragma unroll
        for (int nt = 0; nt < 2; ++nt)
#pragma unroll
            for (int r = 0; r < 8; ++r)
                pb[(r + lh8) * 32 + nt * 16 + ln] = (_Float16)pf[nt][r];

        asm volatile("s_wait_dscnt 0" ::: "memory");   // wave-internal LDS transpose

        v16h pa = load_tile16_lds(pb + ln * 32 + lh8);

        // ---- O += P (16x32) x V (32x64) ----
#pragma unroll
        for (int t = 0; t < 4; ++t) {
            const _Float16* vcol = Vt + ((size_t)h * DHEAD + t * 16 + ln) * S_LEN + j0 + lh8;
            v16h vb = load_tile16(vcol);
            accO[t] = __builtin_amdgcn_wmma_f32_16x16x32_f16(false, pa, false, vb, (short)0, accO[t], false, false);
        }
    }

    // ---- normalize and store O as f16 [S][H*64] ----
    float inv[8];
#pragma unroll
    for (int r = 0; r < 8; ++r) inv[r] = 1.0f / lrun[r];
#pragma unroll
    for (int t = 0; t < 4; ++t)
#pragma unroll
        for (int r = 0; r < 8; ++r)
            Oh[(size_t)(s0 + r + lh8) * (NH * DHEAD) + h * DHEAD + t * 16 + ln] =
                (_Float16)(accO[t][r] * inv[r]);
}

// ---------------- output projection ----------------
// grid (64, 8), block 128.  wave -> 16x64 output tile.  out f32 [S][512]
__global__ __launch_bounds__(128) void out_proj_kernel(
    const _Float16* __restrict__ Oh, const _Float16* __restrict__ Wot,
    float* __restrict__ out) {

    const int lane = threadIdx.x & 31;
    const int wave = threadIdx.x >> 5;
    const int stile = blockIdx.x * 4 + wave;
    const int n0 = blockIdx.y * 64;
    const int s0 = stile * 16;
    const int ln = lane & 15;
    const int lh8 = (lane >> 4) * 8;

    v8f acc[4] = {};
    const _Float16* arow = Oh + (size_t)(s0 + ln) * 512;

    for (int kb = 0; kb < 512; kb += 32) {
        v16h a = load_tile16(arow + kb + lh8);
#pragma unroll
        for (int t = 0; t < 4; ++t) {
            v16h b = load_tile16(Wot + (size_t)(n0 + t * 16 + ln) * 512 + kb + lh8);
            acc[t] = __builtin_amdgcn_wmma_f32_16x16x32_f16(false, a, false, b, (short)0, acc[t], false, false);
        }
    }
#pragma unroll
    for (int t = 0; t < 4; ++t)
#pragma unroll
        for (int r = 0; r < 8; ++r)
            out[(size_t)(s0 + r + lh8) * 512 + n0 + t * 16 + ln] = acc[t][r];
}

// ---------------- host launcher ----------------

extern "C" void kernel_launch(void* const* d_in, const int* in_sizes, int n_in,
                              void* d_out, int out_size, void* d_ws, size_t ws_size,
                              hipStream_t stream) {
    const float* x  = (const float*)d_in[0];   // [1,4096,512]
    const float* Wq = (const float*)d_in[1];   // [8,512,64]
    const float* Wk = (const float*)d_in[2];
    const float* Wv = (const float*)d_in[3];
    const float* Wo = (const float*)d_in[4];   // [8,64,512]
    float* out = (float*)d_out;                // [1,4096,512]

    _Float16* wsh = (_Float16*)d_ws;
    size_t off = 0;
    _Float16* xh  = wsh + off; off += (size_t)S_LEN * D_IN;        // 2M halfs
    _Float16* Wqt = wsh + off; off += (size_t)NH * DHEAD * D_IN;   // 256K
    _Float16* Wkt = wsh + off; off += (size_t)NH * DHEAD * D_IN;
    _Float16* Wvt = wsh + off; off += (size_t)NH * DHEAD * D_IN;
    _Float16* Wot = wsh + off; off += (size_t)D_OUT * (NH * DHEAD);
    _Float16* Qh  = wsh + off; off += (size_t)NH * S_LEN * DHEAD;  // 2M
    _Float16* Kh  = wsh + off; off += (size_t)NH * S_LEN * DHEAD;
    _Float16* Vt  = wsh + off; off += (size_t)NH * DHEAD * S_LEN;
    _Float16* Oh  = wsh + off; off += (size_t)S_LEN * (NH * DHEAD);

    const int nx = S_LEN * D_IN;
    cvt_x_kernel<<<nx / 256, 256, 0, stream>>>(x, xh, nx);
    cvt_w_kernel<<<(NH * D_IN * DHEAD) / 256, 256, 0, stream>>>(Wq, Wqt, 0.125f); // fold 1/sqrt(64)
    cvt_w_kernel<<<(NH * D_IN * DHEAD) / 256, 256, 0, stream>>>(Wk, Wkt, 1.0f);
    cvt_w_kernel<<<(NH * D_IN * DHEAD) / 256, 256, 0, stream>>>(Wv, Wvt, 1.0f);
    cvt_wo_kernel<<<(D_OUT * NH * DHEAD) / 256, 256, 0, stream>>>(Wo, Wot);

    qkv_proj_kernel<<<dim3(64, 8), 128, 0, stream>>>(xh, Wqt, Wkt, Wvt, Qh, Kh, Vt);
    attn_kernel<<<dim3(64, 8), 128, 0, stream>>>(Qh, Kh, Vt, Oh);
    out_proj_kernel<<<dim3(64, 8), 128, 0, stream>>>(Oh, Wot, out);
}
